// LinearEmbed_59794534695066
// MI455X (gfx1250) — matrix operand order repaired and checked by reading, not compiled, not run on hardware
//
#include <hip/hip_runtime.h>
#include <hip/hip_bf16.h>

typedef __attribute__((ext_vector_type(16))) _Float16 v16h;
typedef __attribute__((ext_vector_type(8)))  float    v8f;

#define HID 64
#define NTOT 4096
#define ETOT 32768
#define NN 128
#define BSZ 32
#define EPG 1024

// ---------------- WMMA fragment loaders (wave32, per ISA §7.12.2) -------------
// A operand: 16x32 f16 tile, rows m0..m0+15, k-chunk kc (32 wide).
// lane l: row = m0 + (l&15); holds k = kb..kb+7 (elems 0..7) and kb+16..kb+23
// (elems 8..15), kb = kc*32 + 8*(l>>4).  Two contiguous 16B LDS reads.
__device__ __forceinline__ v16h frag_a(const _Float16* s, int ldk, int m0, int kc, int lane) {
    int r  = m0 + (lane & 15);
    int kb = kc * 32 + ((lane >> 4) << 3);
    const _Float16* p = s + r * ldk + kb;
    v16h a;
    uint4* ap = reinterpret_cast<uint4*>(&a);
    ap[0] = *reinterpret_cast<const uint4*>(p);
    ap[1] = *reinterpret_cast<const uint4*>(p + 16);
    return a;
}
// B operand from WT layout sW[j][k] (so effective B[k][j] = sW[j][k]).
// lane l: j = jt*16 + (l&15); holds k = kb..kb+7 (elems 0..7) and kb+16..kb+23 (elems 8..15), kb = kc*32 + 8*(l>>4).
__device__ __forceinline__ v16h frag_b(const _Float16* s, int ldk, int jt, int kc, int lane) {
    int j  = jt * 16 + (lane & 15);
    int kb = kc * 32 + ((lane >> 4) << 3);
    const _Float16* p = s + j * ldk + kb;
    v16h b;
    uint4* bp = reinterpret_cast<uint4*>(&b);
    bp[0] = *reinterpret_cast<const uint4*>(p);
    bp[1] = *reinterpret_cast<const uint4*>(p + 16);
    return b;
}

// ---------------- weight prep: f32 [k][j] -> f16 WT [j][k], 6 matrices -------
__global__ void k_prep_w(const float* __restrict__ be_w1, const float* __restrict__ be_w2,
                         const float* __restrict__ nn_w1, const float* __restrict__ nn_w2,
                         const float* __restrict__ m_w1,  _Float16* __restrict__ wpack) {
    for (int idx = threadIdx.x; idx < 4096; idx += 256) {
        int j = idx >> 6, k = idx & 63;
        wpack[0 * 4096 + idx] = (_Float16)be_w1[k * 64 + j];
        wpack[1 * 4096 + idx] = (_Float16)be_w2[k * 64 + j];
        wpack[2 * 4096 + idx] = (_Float16)nn_w1[k * 64 + j];
        wpack[3 * 4096 + idx] = (_Float16)nn_w2[k * 64 + j];
        wpack[4 * 4096 + idx] = (_Float16)m_w1[k * 64 + j];          // W1aT (rows 0..63)
        wpack[5 * 4096 + idx] = (_Float16)m_w1[(64 + k) * 64 + j];   // W1bT (rows 64..127)
    }
}

// ---------------- atom encoder: h = x@W_atom + b ; also zero agg -------------
__global__ void k_atom(const float* __restrict__ x, const float* __restrict__ W,
                       const float* __restrict__ b, float* __restrict__ h,
                       float* __restrict__ agg) {
    int t = blockIdx.x * 256 + threadIdx.x;        // 4096*64 threads
    int row = t >> 6, col = t & 63;
    float acc = b[col];
#pragma unroll
    for (int k = 0; k < 32; ++k) acc += x[row * 32 + k] * W[k * 64 + col];
    h[t] = acc;
    agg[t] = 0.0f;
}

// ---------------- bond encoder: e16 = f16(edge_attr@W_bond + b) --------------
__global__ void k_bond(const float* __restrict__ ea, const float* __restrict__ W,
                       const float* __restrict__ b, _Float16* __restrict__ e16) {
    int t = blockIdx.x * 256 + threadIdx.x;        // 32768*64 threads
    int row = t >> 6, col = t & 63;
    float acc = b[col];
#pragma unroll
    for (int k = 0; k < 16; ++k) acc += ea[row * 16 + k] * W[k * 64 + col];
    e16[t] = (_Float16)acc;
}

// ---------------- generic WMMA linear: Y = act(X @ W + bias) -----------------
// X: [gridDim.x*128][64] f16 row-major. WT: [64][64] f16 (transposed weights).
// Compile-time specialization: no runtime branches in the epilogue.
template <int RELU, int HAS_BIAS, int OUT_F32, int OUT_F16>
__global__ void k_wmma_linear(const _Float16* __restrict__ X, const _Float16* __restrict__ WT,
                              const float* __restrict__ bias, float* __restrict__ Yf,
                              _Float16* __restrict__ Yh) {
    __shared__ __align__(16) _Float16 sX[128 * 64];
    __shared__ __align__(16) _Float16 sW[64 * 64];
    int t = threadIdx.x;
    int row0 = blockIdx.x * 128;
    {
        const uint4* g = reinterpret_cast<const uint4*>(X + row0 * 64);
        uint4* d = reinterpret_cast<uint4*>(sX);
#pragma unroll
        for (int q = 0; q < 4; ++q) d[t + 256 * q] = g[t + 256 * q];
        const uint4* gw = reinterpret_cast<const uint4*>(WT);
        uint4* dw = reinterpret_cast<uint4*>(sW);
#pragma unroll
        for (int q = 0; q < 2; ++q) dw[t + 256 * q] = gw[t + 256 * q];
    }
    __syncthreads();
    int wave = t >> 5, lane = t & 31;
    v16h a0 = frag_a(sX, 64, wave * 16, 0, lane);
    v16h a1 = frag_a(sX, 64, wave * 16, 1, lane);
#pragma unroll
    for (int jt = 0; jt < 4; ++jt) {
        v16h b0 = frag_b(sW, 64, jt, 0, lane);
        v16h b1 = frag_b(sW, 64, jt, 1, lane);
        v8f c = {};
        c = __builtin_amdgcn_wmma_f32_16x16x32_f16(false, a0, false, b0, (short)0, c, false, false);
        c = __builtin_amdgcn_wmma_f32_16x16x32_f16(false, a1, false, b1, (short)0, c, false, false);
        int col = jt * 16 + (lane & 15);
        float bb = HAS_BIAS ? bias[col] : 0.0f;
#pragma unroll
        for (int i = 0; i < 8; ++i) {
            int row = row0 + wave * 16 + i + ((lane >> 4) << 3);
            float v = c[i] + bb;
            if (RELU) v = fmaxf(v, 0.0f);
            if (OUT_F32) Yf[row * 64 + col] = v;
            if (OUT_F16) Yh[row * 64 + col] = (_Float16)v;
        }
    }
}

// --------- fused: ee = X@WT + bias ; msg = relu(h[src]+ee) ; agg[dst] += msg --
__global__ void k_wmma_msg(const _Float16* __restrict__ X, const _Float16* __restrict__ WT,
                           const float* __restrict__ bias, const float* __restrict__ h,
                           const int* __restrict__ ei, float* __restrict__ agg) {
    __shared__ __align__(16) _Float16 sX[128 * 64];
    __shared__ __align__(16) _Float16 sW[64 * 64];
    int t = threadIdx.x;
    int row0 = blockIdx.x * 128;
    {
        const uint4* g = reinterpret_cast<const uint4*>(X + row0 * 64);
        uint4* d = reinterpret_cast<uint4*>(sX);
#pragma unroll
        for (int q = 0; q < 4; ++q) d[t + 256 * q] = g[t + 256 * q];
        const uint4* gw = reinterpret_cast<const uint4*>(WT);
        uint4* dw = reinterpret_cast<uint4*>(sW);
#pragma unroll
        for (int q = 0; q < 2; ++q) dw[t + 256 * q] = gw[t + 256 * q];
    }
    __syncthreads();
    int wave = t >> 5, lane = t & 31;
    v16h a0 = frag_a(sX, 64, wave * 16, 0, lane);
    v16h a1 = frag_a(sX, 64, wave * 16, 1, lane);
#pragma unroll
    for (int jt = 0; jt < 4; ++jt) {
        v16h b0 = frag_b(sW, 64, jt, 0, lane);
        v16h b1 = frag_b(sW, 64, jt, 1, lane);
        v8f c = {};
        c = __builtin_amdgcn_wmma_f32_16x16x32_f16(false, a0, false, b0, (short)0, c, false, false);
        c = __builtin_amdgcn_wmma_f32_16x16x32_f16(false, a1, false, b1, (short)0, c, false, false);
        int col = jt * 16 + (lane & 15);
        float bb = bias[col];
#pragma unroll
        for (int i = 0; i < 8; ++i) {
            int eid = row0 + wave * 16 + i + ((lane >> 4) << 3);
            float ee = c[i] + bb;
            int s = ei[eid];
            int d = ei[ETOT + eid];
            float m = fmaxf(h[s * 64 + col] + ee, 0.0f);
            atomicAdd(&agg[d * 64 + col], m);
        }
    }
}

// ---------------- u16 = f16(h + agg) ----------------------------------------
__global__ void k_addh(const float* __restrict__ h, const float* __restrict__ agg,
                       _Float16* __restrict__ u16) {
    int t = blockIdx.x * 256 + threadIdx.x;
    u16[t] = (_Float16)(h[t] + agg[t]);
}

// ---------------- pairwise scorer: one block per (b, n) ----------------------
// Z[m,j] = sum_k L[b,m,k] * ln[k] * W1a[k,j]  (WMMA)  + b1[j] + edge(ec) terms
// out[b,n,m] = sum_j relu(Z[m,j]) * w2[j] + b2
__global__ void k_pair(const _Float16* __restrict__ lg16, const _Float16* __restrict__ W1aT,
                       const float* __restrict__ ec, const float* __restrict__ b1,
                       const float* __restrict__ w2, const float* __restrict__ b2p,
                       const int* __restrict__ ei, float* __restrict__ out) {
    __shared__ __align__(16) _Float16 sL[128 * 64];   // logits of graph b
    __shared__ __align__(16) _Float16 sA[64 * 64];    // (ln ⊙ W1a)^T  [j][k]
    __shared__ float sZ[128 * 64];
    __shared__ _Float16 sln[64];
    __shared__ float sb1[64], sw2[64];
    __shared__ int elist[64];
    __shared__ int ecount;

    int b = blockIdx.x >> 7;
    int n = blockIdx.x & 127;
    int t = threadIdx.x;

    { // load graph logits tile + per-row vectors
        const uint4* g = reinterpret_cast<const uint4*>(lg16 + b * NN * 64);
        uint4* d = reinterpret_cast<uint4*>(sL);
#pragma unroll
        for (int q = 0; q < 4; ++q) d[t + 256 * q] = g[t + 256 * q];
    }
    if (t < 64) {
        sln[t] = lg16[(b * NN + n) * 64 + t];
        sb1[t] = b1[t];
        sw2[t] = w2[t];
    }
    if (t == 0) ecount = 0;
    __syncthreads();

    { // build sA[j][k] = ln[k] * W1aT[j][k] ; init sZ with bias ; scan edges
        int flat = t * 16;
        int k0 = flat & 63;
#pragma unroll
        for (int q = 0; q < 16; ++q)
            sA[flat + q] = sln[k0 + q] * W1aT[flat + q];
        int f2 = t * 32;
#pragma unroll
        for (int q = 0; q < 32; ++q)
            sZ[f2 + q] = sb1[(f2 + q) & 63];
#pragma unroll
        for (int q = 0; q < 4; ++q) {
            int eid = b * EPG + t * 4 + q;
            if (ei[eid] == b * NN + n) {
                int p = atomicAdd(&ecount, 1);
                if (p < 64) elist[p] = eid;
            }
        }
    }
    __syncthreads();

    if (t < 64) { // sparse add of ec rows into Z (thread t owns column t)
        int cnt = ecount < 64 ? ecount : 64;
        for (int i = 0; i < cnt; ++i) {
            int eid = elist[i];
            int dl = ei[ETOT + eid] - b * NN;
            if ((unsigned)dl < (unsigned)NN)
                sZ[dl * 64 + t] += ec[eid * 64 + t];
        }
    }
    __syncthreads();

    { // WMMA: Z += L_b @ sA^T
        int wave = t >> 5, lane = t & 31;
        v16h a0 = frag_a(sL, 64, wave * 16, 0, lane);
        v16h a1 = frag_a(sL, 64, wave * 16, 1, lane);
#pragma unroll
        for (int jt = 0; jt < 4; ++jt) {
            v16h bb0 = frag_b(sA, 64, jt, 0, lane);
            v16h bb1 = frag_b(sA, 64, jt, 1, lane);
            v8f c = {};
            c = __builtin_amdgcn_wmma_f32_16x16x32_f16(false, a0, false, bb0, (short)0, c, false, false);
            c = __builtin_amdgcn_wmma_f32_16x16x32_f16(false, a1, false, bb1, (short)0, c, false, false);
            int col = jt * 16 + (lane & 15);
#pragma unroll
            for (int i = 0; i < 8; ++i) {
                int row = wave * 16 + i + ((lane >> 4) << 3);
                sZ[row * 64 + col] += c[i];
            }
        }
    }
    __syncthreads();

    if (t < 128) { // relu + dot(w2) + b2
        float acc = b2p[0];
#pragma unroll 8
        for (int j = 0; j < 64; ++j)
            acc += fmaxf(sZ[t * 64 + j], 0.0f) * sw2[j];
        out[b * (NN * NN) + n * NN + t] = acc;
    }
}

// ---------------- mask = ones ------------------------------------------------
__global__ void k_ones(float* __restrict__ p) {
    p[blockIdx.x * 256 + threadIdx.x] = 1.0f;
}

extern "C" void kernel_launch(void* const* d_in, const int* in_sizes, int n_in,
                              void* d_out, int out_size, void* d_ws, size_t ws_size,
                              hipStream_t stream) {
    const float* x        = (const float*)d_in[0];
    const float* edge_attr= (const float*)d_in[1];
    const int*   ei       = (const int*)d_in[2];
    const float* W_atom   = (const float*)d_in[4];
    const float* b_atom   = (const float*)d_in[5];
    const float* W_bond   = (const float*)d_in[6];
    const float* b_bond   = (const float*)d_in[7];
    const float* be_w1    = (const float*)d_in[8];
    const float* be_b1    = (const float*)d_in[9];
    const float* be_w2    = (const float*)d_in[10];
    const float* be_b2    = (const float*)d_in[11];
    const float* nn_w1    = (const float*)d_in[12];
    const float* nn_b1    = (const float*)d_in[13];
    const float* nn_w2    = (const float*)d_in[14];
    const float* nn_b2    = (const float*)d_in[15];
    const float* m_w1     = (const float*)d_in[16];
    const float* m_b1     = (const float*)d_in[17];
    const float* m_w2     = (const float*)d_in[18];
    const float* m_b2     = (const float*)d_in[19];

    char* ws = (char*)d_ws;
    float*    h    = (float*)   (ws + 0);                               // 1 MB
    float*    agg  = (float*)   (ws + (1u << 20));                      // 1 MB
    _Float16* e16  = (_Float16*)(ws + 2u * (1u << 20));                 // 4 MB
    _Float16* t16  = (_Float16*)(ws + 6u * (1u << 20));                 // 4 MB
    float*    ec   = (float*)   (ws + 10u * (1u << 20));                // 8 MB
    _Float16* u16  = (_Float16*)(ws + 18u * (1u << 20));                // 0.5 MB
    _Float16* t216 = (_Float16*)(ws + 18u * (1u << 20) + 512u * 1024);  // 0.5 MB
    _Float16* lg16 = (_Float16*)(ws + 18u * (1u << 20) + 1024u * 1024); // 0.5 MB
    _Float16* wpk  = (_Float16*)(ws + 18u * (1u << 20) + 1536u * 1024); // 48 KB

    float* out_emb  = (float*)d_out;
    float* out_mask = out_emb + BSZ * NN * NN;

    k_prep_w<<<1, 256, 0, stream>>>(be_w1, be_w2, nn_w1, nn_w2, m_w1, wpk);
    k_atom  <<<NTOT * 64 / 256, 256, 0, stream>>>(x, W_atom, b_atom, h, agg);
    k_bond  <<<ETOT * 64 / 256, 256, 0, stream>>>(edge_attr, W_bond, b_bond, e16);
    // bond-encoder MLP layer 1: t16 = relu(e16 @ be_w1 + be_b1) -> f16
    k_wmma_linear<1, 1, 0, 1><<<ETOT / 128, 256, 0, stream>>>(e16, wpk + 0 * 4096, be_b1, nullptr, t16);
    // layer 2 fused with message + scatter-add into agg
    k_wmma_msg<<<ETOT / 128, 256, 0, stream>>>(t16, wpk + 1 * 4096, be_b2, h, ei, agg);
    // node update MLP
    k_addh<<<NTOT * 64 / 256, 256, 0, stream>>>(h, agg, u16);
    k_wmma_linear<1, 1, 0, 1><<<NTOT / 128, 256, 0, stream>>>(u16, wpk + 2 * 4096, nn_b1, nullptr, t216);
    k_wmma_linear<0, 1, 0, 1><<<NTOT / 128, 256, 0, stream>>>(t216, wpk + 3 * 4096, nn_b2, nullptr, lg16);
    // ec = e @ W1b (edge half of final MLP layer 1) -> f32
    k_wmma_linear<0, 0, 1, 0><<<ETOT / 128, 256, 0, stream>>>(e16, wpk + 5 * 4096, nullptr, ec, nullptr);
    // pairwise scorer
    k_pair<<<BSZ * NN, 256, 0, stream>>>(lg16, wpk + 4 * 4096, ec, m_b1, m_w2, m_b2, ei, out_emb);
    k_ones<<<BSZ * NN * NN / 256, 256, 0, stream>>>(out_mask);
}
